// ClusteringLoss_62053687492907
// MI455X (gfx1250) — matrix operand order repaired and checked by reading, not compiled, not run on hardware
//
#include <hip/hip_runtime.h>
#include <hip/hip_bf16.h>

// KMeans (K=32, D=64, 10 iters) + min-dist loss for MI455X / gfx1250.
// GEMM part runs on V_WMMA_F32_16X16X32_BF16 with fp32->bf16 hi/lo split
// (3 products ~= fp32 accuracy). X is L2-resident (128MB < 192MB L2).

typedef __attribute__((ext_vector_type(16))) __bf16 v16bf;
typedef __attribute__((ext_vector_type(8)))  float  v8f;

#define KC 32          // clusters
#define DD 64          // feature dim
#define NITER 10
#define TPB 256        // 8 waves of 32
#define WPB 8

static __device__ __forceinline__ v8f wmma_bf16(v16bf a, v16bf b, v8f c) {
  // D = A(16x32 bf16) * B(32x16 bf16) + C(16x16 f32)
  return __builtin_amdgcn_wmma_f32_16x16x32_bf16(false, a, false, b, (short)0, c,
                                                 false, false);
}

static __device__ __forceinline__ float swap16_f(float v) {
  // group-of-32 ds_swizzle, SWAPX16: xor=0x10,or=0,and=0x1f -> 0x401F
  return __int_as_float(__builtin_amdgcn_ds_swizzle(__float_as_int(v), 0x401F));
}
static __device__ __forceinline__ int swap16_i(int v) {
  return __builtin_amdgcn_ds_swizzle(v, 0x401F);
}

// ---------------------------------------------------------------------------
// Centroid prep: init (strided rows of Z) or update (csum/cnt), then emit
// fp32 centroids, bf16 hi/lo split, squared norms; zero accumulators.
// ---------------------------------------------------------------------------
__global__ __launch_bounds__(TPB)
void kmeans_prep(const float* __restrict__ Z, float* __restrict__ cent,
                 __bf16* __restrict__ chi, __bf16* __restrict__ clo,
                 float* __restrict__ cnorm, float* __restrict__ csum,
                 float* __restrict__ cnt, float* __restrict__ loss,
                 int N, int do_init) {
  __shared__ float cn_lds[KC];
  const int tid = threadIdx.x;
  if (tid < KC) cn_lds[tid] = 0.0f;
  __syncthreads();

  const int stride = N / KC;  // deterministic "sampled" init rows
  for (int e = tid; e < KC * DD; e += TPB) {
    const int j = e >> 6;
    const int k = e & 63;
    float v;
    if (do_init) {
      v = Z[(size_t)j * (size_t)stride * DD + k];
    } else {
      const float c = cnt[j];
      v = (c > 0.0f) ? (csum[e] / c) : 0.0f;   // empty cluster -> 0 (as ref)
    }
    cent[e] = v;
    const __bf16 hi = (__bf16)v;
    chi[e] = hi;
    clo[e] = (__bf16)(v - (float)hi);
    atomicAdd(&cn_lds[j], v * v);
    csum[e] = 0.0f;                            // reset for next pass
  }
  __syncthreads();
  if (tid < KC) { cnorm[tid] = cn_lds[tid]; cnt[tid] = 0.0f; }
  if (tid == 0) loss[0] = 0.0f;
}

// ---------------------------------------------------------------------------
// Assignment pass: per wave, 16 rows/tile. D = C(32x64) * Xtile^T(64x16)
// via 12 bf16 WMMAs (2 ktiles x 2 centroid-halves x 3 hi/lo products).
// ---------------------------------------------------------------------------
__global__ __launch_bounds__(TPB)
void kmeans_assign(const float* __restrict__ Z, int N, int ntiles,
                   const __bf16* __restrict__ chi, const __bf16* __restrict__ clo,
                   const float* __restrict__ cnorm,
                   float* __restrict__ g_csum, float* __restrict__ g_cnt,
                   float* __restrict__ g_loss, float* __restrict__ cl_out,
                   int do_accum, int do_loss) {
  __shared__ float csum_lds[KC * (DD + 1)];    // +1 pad: kill 16-way bank conflicts
  __shared__ float cnt_lds[KC];
  __shared__ float loss_lds;

  const int tid  = threadIdx.x;
  const int lane = tid & 31;
  const int wid  = tid >> 5;
  const int n    = lane & 15;      // column in D  = x-row within tile
  const int h    = lane >> 4;      // lane half    = K/M sub-stripe select

  for (int e = tid; e < KC * (DD + 1); e += TPB) csum_lds[e] = 0.0f;
  if (tid < KC) cnt_lds[tid] = 0.0f;
  if (tid == 0) loss_lds = 0.0f;
  __syncthreads();

  // ---- A fragments (centroids), resident in VGPRs for the whole kernel.
  // 16-bit A 16x32 layout: lane m+16h, element t -> K=h*16+t (contig, matches B)
  v16bf aHi[2][2], aLo[2][2];      // [ktile][centroid-half]
#pragma unroll
  for (int mt = 0; mt < 2; ++mt) {
#pragma unroll
    for (int kt = 0; kt < 2; ++kt) {
      const __bf16* bh = chi + ((size_t)(mt * 16 + n)) * DD + kt * 32;
      const __bf16* bl = clo + ((size_t)(mt * 16 + n)) * DD + kt * 32;
#pragma unroll
      for (int t = 0; t < 8; ++t) {
        aHi[kt][mt][t]     = bh[h * 16 + t];
        aHi[kt][mt][8 + t] = bh[h * 16 + 8 + t];
        aLo[kt][mt][t]     = bl[h * 16 + t];
        aLo[kt][mt][8 + t] = bl[h * 16 + 8 + t];
      }
    }
  }
  // ||c_j||^2 for the 16 centroids this lane produces (8 per D-tile).
  float cn0[8], cn1[8];
#pragma unroll
  for (int r = 0; r < 8; ++r) {
    cn0[r] = cnorm[8 * h + r];
    cn1[r] = cnorm[16 + 8 * h + r];
  }

  const int gwave  = blockIdx.x * WPB + wid;
  const int nwaves = gridDim.x * WPB;

  for (int tile = gwave; tile < ntiles; tile += nwaves) {  // wave-uniform bound
    const int rowi   = tile * 16 + n;
    const bool valid = rowi < N;
    const int row    = valid ? rowi : (N - 1);
    const float* xr  = Z + (size_t)row * DD;

    float fX[2][16];
    float sq = 0.0f;
    v8f acc0 = {0.f, 0.f, 0.f, 0.f, 0.f, 0.f, 0.f, 0.f};
    v8f acc1 = {0.f, 0.f, 0.f, 0.f, 0.f, 0.f, 0.f, 0.f};

#pragma unroll
    for (int kt = 0; kt < 2; ++kt) {
      // B 32x16 layout: lane n(+16h) holds K = kt*32 + h*16 + t, t=0..15 (contig)
      const float4* p = (const float4*)(xr + kt * 32 + h * 16);
#pragma unroll
      for (int v4 = 0; v4 < 4; ++v4) {
        const float4 q = p[v4];
        fX[kt][v4 * 4 + 0] = q.x;
        fX[kt][v4 * 4 + 1] = q.y;
        fX[kt][v4 * 4 + 2] = q.z;
        fX[kt][v4 * 4 + 3] = q.w;
      }
      v16bf bHi, bLo;
#pragma unroll
      for (int t = 0; t < 16; ++t) {
        const float f  = fX[kt][t];
        const __bf16 hi = (__bf16)f;
        bHi[t] = hi;
        bLo[t] = (__bf16)(f - (float)hi);
        sq += f * f;
      }
      // split-precision dot: hi*hi + lo*hi + hi*lo
      acc0 = wmma_bf16(aHi[kt][0], bHi, acc0);
      acc0 = wmma_bf16(aLo[kt][0], bHi, acc0);
      acc0 = wmma_bf16(aHi[kt][0], bLo, acc0);
      acc1 = wmma_bf16(aHi[kt][1], bHi, acc1);
      acc1 = wmma_bf16(aLo[kt][1], bHi, acc1);
      acc1 = wmma_bf16(aHi[kt][1], bLo, acc1);
    }

    // argmin over this lane's 16 centroids of (||c||^2 - 2*dot), j ascending
    float best = 3.0e38f;
    int bestj = 0;
#pragma unroll
    for (int r = 0; r < 8; ++r) {
      const float v = cn0[r] - 2.0f * acc0[r];
      if (v < best) { best = v; bestj = 8 * h + r; }
    }
#pragma unroll
    for (int r = 0; r < 8; ++r) {
      const float v = cn1[r] - 2.0f * acc1[r];
      if (v < best) { best = v; bestj = 16 + 8 * h + r; }
    }
    // combine the two lane-halves (lane n <-> lane n+16); smaller j on ties
    const float ob = swap16_f(best);
    const int   oj = swap16_i(bestj);
    if (ob < best || (ob == best && oj < bestj)) { best = ob; bestj = oj; }
    const float xnorm = sq + swap16_f(sq);

    if (do_accum && valid) {
      // segment-sum: each half-lane adds the 32 x-values it already holds
#pragma unroll
      for (int kt = 0; kt < 2; ++kt) {
#pragma unroll
        for (int t = 0; t < 16; ++t)
          atomicAdd(&csum_lds[bestj * (DD + 1) + kt * 32 + h * 16 + t], fX[kt][t]);
      }
      if (h == 0) atomicAdd(&cnt_lds[bestj], 1.0f);
    }
    if (do_loss && valid && h == 0) atomicAdd(&loss_lds, xnorm + best);
    if (cl_out && valid && h == 0) cl_out[rowi] = (float)bestj;
  }

  __syncthreads();
  if (do_accum) {
    for (int e = tid; e < KC * DD; e += TPB) {
      const int j = e >> 6, k = e & 63;
      atomicAdd(&g_csum[e], csum_lds[j * (DD + 1) + k]);
    }
    if (tid < KC) atomicAdd(&g_cnt[tid], cnt_lds[tid]);
  }
  if (do_loss && tid == 0) atomicAdd(g_loss, loss_lds);
}

__global__ void kmeans_finish(const float* __restrict__ loss,
                              float* __restrict__ out, float invN) {
  out[0] = loss[0] * invN;
}

// ---------------------------------------------------------------------------
extern "C" void kernel_launch(void* const* d_in, const int* in_sizes, int n_in,
                              void* d_out, int out_size, void* d_ws, size_t ws_size,
                              hipStream_t stream) {
  const float* Z = (const float*)d_in[0];
  const int N = in_sizes[0] / DD;            // 500000
  const int ntiles = (N + 15) / 16;

  float* out = (float*)d_out;                // [loss, cl(0..N-1)]
  float* cl_out = out + 1;

  // workspace carve-up (all 16B-aligned-ish; ~25 KB total)
  float* csum  = (float*)d_ws;               // 2048
  float* cnt   = csum + KC * DD;             // 32
  float* cent  = cnt + KC;                   // 2048
  float* cnorm = cent + KC * DD;             // 32
  float* loss  = cnorm + KC;                 // 1 (+pad)
  __bf16* chi  = (__bf16*)(loss + 8);        // 2048 bf16
  __bf16* clo  = chi + KC * DD;              // 2048 bf16

  int blocks = (ntiles + WPB - 1) / WPB;
  if (blocks > 1024) blocks = 1024;
  if (blocks < 1) blocks = 1;

  // init centroids (deterministic strided sample) + zero accumulators
  kmeans_prep<<<1, TPB, 0, stream>>>(Z, cent, chi, clo, cnorm, csum, cnt, loss, N, 1);

  for (int it = 0; it < NITER; ++it) {
    float* cl = (it == NITER - 1) ? cl_out : nullptr;  // cls[-1] uses pre-update c
    kmeans_assign<<<blocks, TPB, 0, stream>>>(Z, N, ntiles, chi, clo, cnorm,
                                              csum, cnt, loss, cl,
                                              /*do_accum=*/1, /*do_loss=*/0);
    kmeans_prep<<<1, TPB, 0, stream>>>(Z, cent, chi, clo, cnorm, csum, cnt, loss, N, 0);
  }
  // loss pass with final centroids (after 10 updates)
  kmeans_assign<<<blocks, TPB, 0, stream>>>(Z, N, ntiles, chi, clo, cnorm,
                                            csum, cnt, loss, nullptr,
                                            /*do_accum=*/0, /*do_loss=*/1);
  kmeans_finish<<<1, 1, 0, stream>>>(loss, out, 1.0f / (float)N);
}
